// SE3MixAttention_7378753815040
// MI455X (gfx1250) — matrix operand order repaired and checked by reading, compiled, not run on hardware
//
#include <hip/hip_runtime.h>
#include <hip/hip_fp16.h>

typedef __attribute__((ext_vector_type(16))) _Float16 v16h;
typedef __attribute__((ext_vector_type(8)))  _Float16 v8h;
typedef __attribute__((ext_vector_type(4)))  _Float16 v4h;
typedef __attribute__((ext_vector_type(8)))  float    v8f;

#define NN 50000
#define EE 800000
#define GG 200

__device__ __forceinline__ float siluf(float x) { return x / (1.0f + __expf(-x)); }

// ---- WMMA fragment helpers (CDNA5 16x16x32 f16 layouts, wave32) ----

// A: 16x32 f16 tile from LDS (row-major, stride ldk halves), k offset k0.
// lanes 0-15: rows 0-15, K = {kb..kb+7} U {kb+16..kb+23}, kb=0; lanes 16-31: kb=8.
// Both 8-half runs are 16B aligned -> 2x ds_load_b128.
__device__ __forceinline__ v16h load_a_frag(const _Float16* lds, int ldk, int k0, int lane) {
  int m  = lane & 15;
  int kb = (lane & 16) ? 8 : 0;
  const _Float16* row = lds + m * ldk + k0 + kb;
  v8h lo = *(const v8h*)(row);
  v8h hi = *(const v8h*)(row + 16);
  return __builtin_shufflevector(lo, hi, 0, 1, 2, 3, 4, 5, 6, 7, 8, 9, 10, 11, 12, 13, 14, 15);
}

// B: pre-swizzled fragment-major weight; lane's 16 halves contiguous -> 2x global_load_b128.
__device__ __forceinline__ v16h load_b_packed(const _Float16* wp, int frag, int lane) {
  const _Float16* p = wp + ((long)frag * 32 + lane) * 16;
  v8h lo = *(const v8h*)(p);
  v8h hi = *(const v8h*)(p + 8);
  return __builtin_shufflevector(lo, hi, 0, 1, 2, 3, 4, 5, 6, 7, 8, 9, 10, 11, 12, 13, 14, 15);
}

__device__ __forceinline__ v8f wmma16(v16h a, v16h b, v8f c) {
  return __builtin_amdgcn_wmma_f32_16x16x32_f16(false, a, false, b, (short)0, c, false, false);
}

// C/D f32 layout: VGPR v holds row (v + 8*(lane>=16)), col = lane&15.
__device__ __forceinline__ int c_row0(int lane) { return (lane & 16) ? 8 : 0; }

__device__ __forceinline__ void cvt_store4(_Float16* dst, float4 t) {
  v4h o;
  o[0] = (_Float16)t.x; o[1] = (_Float16)t.y; o[2] = (_Float16)t.z; o[3] = (_Float16)t.w;
  *(v4h*)dst = o;   // 8B aligned -> ds_store_b64
}

// stage 16 gathered f32 rows (64 wide) -> f16 LDS tile (stride 64)
__device__ __forceinline__ void stage_rows_f16(const float* g, const int* rows, _Float16* SA, int lane) {
  int r = lane & 15;
  int c0 = (lane & 16) ? 32 : 0;
  const float4* src = (const float4*)(g + (long)rows[r] * 64 + c0);
  _Float16* dst = SA + r * 64 + c0;
#pragma unroll
  for (int j = 0; j < 8; ++j) cvt_store4(dst + 4 * j, src[j]);
}

// stage 16 consecutive f32 rows (64 wide) -> f16 LDS tile (stride 64)
__device__ __forceinline__ void stage_seq_f16(const float* g, long base_row, _Float16* SA, int lane) {
  int r = lane & 15;
  int c0 = (lane & 16) ? 32 : 0;
  const float4* src = (const float4*)(g + (base_row + r) * 64 + c0);
  _Float16* dst = SA + r * 64 + c0;
#pragma unroll
  for (int j = 0; j < 8; ++j) cvt_store4(dst + 4 * j, src[j]);
}

// -------------------- utility kernels --------------------

__global__ void zero_f32(float* p, long n) {
  long i = (long)blockIdx.x * blockDim.x + threadIdx.x;
  long stride = (long)gridDim.x * blockDim.x;
  for (; i < n; i += stride) p[i] = 0.0f;
}

// repack f32 row-major (K+row_off rows x Nout) weight into f16 fragment-major layout:
// dst[((kt*NT + nt)*32 + lane)*16 + e] = w[kt*32 + 8*(lane>=16) + e + 8*(e>=8) + row_off][nt*16 + (lane&15)]
__global__ void pack_b_f16(const float* __restrict__ src, _Float16* __restrict__ dst,
                           int Nout, int row_off, long total) {
  long tid = (long)blockIdx.x * blockDim.x + threadIdx.x;
  if (tid >= total) return;
  int e    = (int)(tid & 15);
  int lane = (int)((tid >> 4) & 31);
  long frag = tid >> 9;
  int NT = Nout >> 4;
  int nt = (int)(frag % NT);
  int kt = (int)(frag / NT);
  int n = nt * 16 + (lane & 15);
  int k = kt * 32 + ((lane & 16) ? 8 : 0) + e + ((e & 8) ? 8 : 0);
  dst[tid] = (_Float16)src[(long)(k + row_off) * Nout + n];
}

__global__ void norm_kernel(const float* __restrict__ X, const int* __restrict__ batch,
                            float* __restrict__ sumn, float* __restrict__ cnt) {
  int i = blockIdx.x * blockDim.x + threadIdx.x;
  if (i >= NN) return;
  float x = X[3 * i], y = X[3 * i + 1], z = X[3 * i + 2];
  float nrm = sqrtf(x * x + y * y + z * z);
  int b = batch[i];
  atomicAdd(&sumn[b], nrm);
  atomicAdd(&cnt[b], 1.0f);
}

__global__ void xn_kernel(const float* __restrict__ X, const int* __restrict__ batch,
                          const float* __restrict__ sumn, const float* __restrict__ cnt,
                          const float* __restrict__ e3w, float* __restrict__ Xn) {
  int i = blockIdx.x * blockDim.x + threadIdx.x;
  if (i >= NN) return;
  int b = batch[i];
  float mean = sumn[b] / fmaxf(cnt[b], 1.0f);
  float s = e3w[0] / (mean + 1e-5f);
  Xn[3 * i + 0] = X[3 * i + 0] * s;
  Xn[3 * i + 1] = X[3 * i + 1] * s;
  Xn[3 * i + 2] = X[3 * i + 2] * s;
}

// -------------------- gate kernel (WMMA) --------------------

struct __align__(16) NodeLds {
  _Float16 SA[16 * 64];
  _Float16 SB[16 * 64];
  float    red[32 * 8];
  float    sval[16];
};

__global__ __launch_bounds__(64) void gate_kernel(
    const float* __restrict__ H, const _Float16* __restrict__ w1p,
    const float* __restrict__ b1, const float* __restrict__ w2,
    const float* __restrict__ b2, float* __restrict__ gate, int ntiles) {
  __shared__ NodeLds L[2];
  int wave = threadIdx.x >> 5, lane = threadIdx.x & 31;
  int tile = blockIdx.x * 2 + wave;
  if (tile >= ntiles) tile = ntiles - 1;   // idempotent duplicate
  NodeLds& S = L[wave];
  long base = (long)tile * 16;

  stage_seq_f16(H, base, S.SA, lane);
  __syncthreads();

  float acc8[8];
#pragma unroll
  for (int v = 0; v < 8; ++v) acc8[v] = 0.0f;
  int col = lane & 15;
#pragma unroll
  for (int ct = 0; ct < 4; ++ct) {
    v8f c = {0.f, 0.f, 0.f, 0.f, 0.f, 0.f, 0.f, 0.f};
#pragma unroll
    for (int kt = 0; kt < 2; ++kt) {
      v16h a = load_a_frag(S.SA, 64, kt * 32, lane);
      v16h b = load_b_packed(w1p, kt * 4 + ct, lane);
      c = wmma16(a, b, c);
    }
    int n = ct * 16 + col;
    float b1n = b1[n], w2n = w2[n];
#pragma unroll
    for (int v = 0; v < 8; ++v) acc8[v] += siluf(c[v] + b1n) * w2n;
  }
#pragma unroll
  for (int v = 0; v < 8; ++v) S.red[lane * 8 + v] = acc8[v];
  __syncthreads();
  if (lane < 16) {
    int m = lane, v = m & 7, half = m >> 3;
    float s = 0.0f;
#pragma unroll
    for (int j = 0; j < 16; ++j) s += S.red[(half * 16 + j) * 8 + v];
    s += b2[0];
    gate[base + m] = 1.0f / (1.0f + __expf(-s));
  }
}

// -------------------- per-edge kernel --------------------

struct __align__(16) EdgeLds {
  _Float16 SA[16 * 64];    // staged A operand (Hdst / Hsrc / ZE)
  _Float16 SB[16 * 128];   // f16 hidden activations (Q hidden 16x64, KV hidden 16x128)
  float    SC[16 * 192];   // cols 0..63 = K, 64..127 = V, 128..191 = Q (f32)
  float    red[32 * 8];
  float    relD[16];
  float    xrn[16 * 3];
  float    sB[16];
  float    salpha[16];
  int      ssrc[16];
  int      sdst[16];
};

__global__ __launch_bounds__(64) void edge_kernel(
    const int* __restrict__ Eidx, const float* __restrict__ H,
    const float* __restrict__ Xn, const float* __restrict__ ZE,
    const _Float16* __restrict__ qw1p, const _Float16* __restrict__ qw2p,
    const _Float16* __restrict__ kvw1p, const _Float16* __restrict__ kvw2p,
    const _Float16* __restrict__ pbw1p,
    const float* __restrict__ kv_w1f,   // original f32 (65x128); row 0 = rel_dist weights
    const float* __restrict__ pb_b1,
    const float* __restrict__ pb_w2, const float* __restrict__ pb_b2,
    const float* __restrict__ qn_w, const float* __restrict__ qn_b,
    const float* __restrict__ kn_w, const float* __restrict__ kn_b,
    const float* __restrict__ gate,
    float* __restrict__ A, float* __restrict__ AX, float* __restrict__ AH) {
  __shared__ EdgeLds L[2];
  int wave = threadIdx.x >> 5, lane = threadIdx.x & 31;
  EdgeLds& S = L[wave];
  long tile = (long)blockIdx.x * 2 + wave;   // E/16 = 50000 tiles exactly
  long base = tile * 16;
  int col = lane & 15;
  int row0 = c_row0(lane);

  // Phase 0: indices, rel geometry
  if (lane < 16) {
    int m = lane;
    long e = base + m;
    int s = Eidx[e], d = Eidx[EE + e];
    S.ssrc[m] = s; S.sdst[m] = d;
    float rx = Xn[s * 3 + 0] - Xn[d * 3 + 0];
    float ry = Xn[s * 3 + 1] - Xn[d * 3 + 1];
    float rz = Xn[s * 3 + 2] - Xn[d * 3 + 2];
    float rd = rx * rx + ry * ry + rz * rz;
    S.relD[m] = rd;
    float inv = 1.0f / (1.0f + sqrtf(rd + 1e-8f));
    S.xrn[m * 3 + 0] = rx * inv; S.xrn[m * 3 + 1] = ry * inv; S.xrn[m * 3 + 2] = rz * inv;
  }
  __syncthreads();

  // Phase 1: stage H[dst]
  stage_rows_f16(H, S.sdst, S.SA, lane);
  __syncthreads();

  // Phase 2: Q hidden = silu(Hdst @ q_w1) -> SB (16x64 f16)
#pragma unroll
  for (int ct = 0; ct < 4; ++ct) {
    v8f c = {0.f, 0.f, 0.f, 0.f, 0.f, 0.f, 0.f, 0.f};
#pragma unroll
    for (int kt = 0; kt < 2; ++kt) {
      v16h a = load_a_frag(S.SA, 64, kt * 32, lane);
      v16h b = load_b_packed(qw1p, kt * 4 + ct, lane);
      c = wmma16(a, b, c);
    }
    int n = ct * 16 + col;
#pragma unroll
    for (int v = 0; v < 8; ++v) S.SB[(row0 + v) * 64 + n] = (_Float16)siluf(c[v]);
  }
  __syncthreads();

  // Phase 3: Q = hidden @ q_w2 -> SC[:,128..191] (f32)
#pragma unroll
  for (int ct = 0; ct < 4; ++ct) {
    v8f c = {0.f, 0.f, 0.f, 0.f, 0.f, 0.f, 0.f, 0.f};
#pragma unroll
    for (int kt = 0; kt < 2; ++kt) {
      v16h a = load_a_frag(S.SB, 64, kt * 32, lane);
      v16h b = load_b_packed(qw2p, kt * 4 + ct, lane);
      c = wmma16(a, b, c);
    }
    int n = ct * 16 + col;
#pragma unroll
    for (int v = 0; v < 8; ++v) S.SC[(row0 + v) * 192 + 128 + n] = c[v];
  }
  __syncthreads();

  // Phase 4: stage H[src]
  stage_rows_f16(H, S.ssrc, S.SA, lane);
  __syncthreads();

  // Phase 5: KV hidden = silu(Hsrc @ kv_w1[1:,:] + relD (x) kv_w1[0,:]) -> SB (16x128 f16)
#pragma unroll
  for (int ct = 0; ct < 8; ++ct) {
    v8f c = {0.f, 0.f, 0.f, 0.f, 0.f, 0.f, 0.f, 0.f};
#pragma unroll
    for (int kt = 0; kt < 2; ++kt) {
      v16h a = load_a_frag(S.SA, 64, kt * 32, lane);
      v16h b = load_b_packed(kvw1p, kt * 8 + ct, lane);
      c = wmma16(a, b, c);
    }
    int n = ct * 16 + col;
    float w0 = kv_w1f[n];   // row 0 of kv_w1 (rel_dist column), f32
#pragma unroll
    for (int v = 0; v < 8; ++v) {
      float x = c[v] + S.relD[row0 + v] * w0;
      S.SB[(row0 + v) * 128 + n] = (_Float16)siluf(x);
    }
  }
  __syncthreads();

  // Phase 6: KV = hidden @ kv_w2 (128x128) -> SC[:,0..127] (f32)
#pragma unroll
  for (int ct = 0; ct < 8; ++ct) {
    v8f c = {0.f, 0.f, 0.f, 0.f, 0.f, 0.f, 0.f, 0.f};
#pragma unroll
    for (int kt = 0; kt < 4; ++kt) {
      v16h a = load_a_frag(S.SB, 128, kt * 32, lane);
      v16h b = load_b_packed(kvw2p, kt * 8 + ct, lane);
      c = wmma16(a, b, c);
    }
    int n = ct * 16 + col;
#pragma unroll
    for (int v = 0; v < 8; ++v) S.SC[(row0 + v) * 192 + n] = c[v];
  }
  __syncthreads();

  // Phase 7: pair-bias B = silu(ZE @ pb_w1 + b1) @ pb_w2 + b2
  {
    int r = lane & 15, c0 = (lane & 16) ? 32 : 0;
    const float4* src = (const float4*)(ZE + (base + r) * 64 + c0);
    _Float16* dst = S.SA + r * 64 + c0;
#pragma unroll
    for (int j = 0; j < 8; ++j) cvt_store4(dst + 4 * j, src[j]);
  }
  __syncthreads();
  {
    float acc8[8];
#pragma unroll
    for (int v = 0; v < 8; ++v) acc8[v] = 0.0f;
#pragma unroll
    for (int ct = 0; ct < 16; ++ct) {
      v8f c = {0.f, 0.f, 0.f, 0.f, 0.f, 0.f, 0.f, 0.f};
#pragma unroll
      for (int kt = 0; kt < 2; ++kt) {
        v16h a = load_a_frag(S.SA, 64, kt * 32, lane);
        v16h b = load_b_packed(pbw1p, kt * 16 + ct, lane);
        c = wmma16(a, b, c);
      }
      int n = ct * 16 + col;
      float b1n = pb_b1[n], w2n = pb_w2[n];
#pragma unroll
      for (int v = 0; v < 8; ++v) acc8[v] += siluf(c[v] + b1n) * w2n;
    }
#pragma unroll
    for (int v = 0; v < 8; ++v) S.red[lane * 8 + v] = acc8[v];
  }
  __syncthreads();
  if (lane < 16) {
    int m = lane, v = m & 7, half = m >> 3;
    float s = 0.0f;
#pragma unroll
    for (int j = 0; j < 16; ++j) s += S.red[(half * 16 + j) * 8 + v];
    S.sB[m] = s + pb_b2[0];
  }
  __syncthreads();

  // Phase 8: layernorm(Q), layernorm(K), scores, alpha
  if (lane < 16) {
    int m = lane;
    const float* qrow = &S.SC[m * 192 + 128];
    const float* krow = &S.SC[m * 192 + 0];
    float mq = 0.f, mk = 0.f;
    for (int k = 0; k < 64; ++k) { mq += qrow[k]; mk += krow[k]; }
    mq *= (1.0f / 64.0f); mk *= (1.0f / 64.0f);
    float vq = 0.f, vk = 0.f;
    for (int k = 0; k < 64; ++k) {
      float a = qrow[k] - mq, b = krow[k] - mk;
      vq += a * a; vk += b * b;
    }
    float rq = rsqrtf(vq * (1.0f / 64.0f) + 1e-5f);
    float rk = rsqrtf(vk * (1.0f / 64.0f) + 1e-5f);
    float dot = 0.f;
    for (int k = 0; k < 64; ++k) {
      float qn = (qrow[k] - mq) * rq * qn_w[k] + qn_b[k];
      float kn = (krow[k] - mk) * rk * kn_w[k] + kn_b[k];
      dot += qn * kn;
    }
    float score = dot * 0.125f + S.sB[m];   // 1/sqrt(64)
    S.salpha[m] = gate[S.ssrc[m]] * score;
  }
  __syncthreads();

  // Phase 9: atomic scatter of A, AH, AX (segment_sum over dst)
  {
    int r = lane & 15, c0 = (lane & 16) ? 32 : 0;
    int d = S.sdst[r];
    float alpha = S.salpha[r];
    const float* vrow = &S.SC[r * 192 + 64];
    const float* hrow = H + (long)d * 64;
    float* Arow = A + (long)d * 64;
    float* AHrow = AH + (long)d * 64;
    for (int j = 0; j < 32; ++j) {
      int cc = c0 + j;
      float att = alpha * vrow[cc];
      atomicAdd(&Arow[cc], att);
      atomicAdd(&AHrow[cc], att * hrow[cc]);
    }
    if (lane < 16) {
#pragma unroll
      for (int k = 0; k < 3; ++k) atomicAdd(&AX[(long)d * 3 + k], alpha * S.xrn[r * 3 + k]);
    }
  }
}

// -------------------- node output kernel (WMMA) --------------------

__global__ __launch_bounds__(64) void nodeout_kernel(
    const float* __restrict__ H, const float* __restrict__ Xn,
    const float* __restrict__ A, const float* __restrict__ AX, const float* __restrict__ AH,
    const _Float16* __restrict__ phixw1p, const float* __restrict__ phix_w2,
    const _Float16* __restrict__ phihw1p, const float* __restrict__ phih_b1,
    const _Float16* __restrict__ phihw2p, const float* __restrict__ phih_b2,
    float* __restrict__ Xout, float* __restrict__ Hout, int ntiles) {
  __shared__ NodeLds L[2];
  int wave = threadIdx.x >> 5, lane = threadIdx.x & 31;
  int tile = blockIdx.x * 2 + wave;
  if (tile >= ntiles) tile = ntiles - 1;   // idempotent duplicate
  NodeLds& S = L[wave];
  long base = (long)tile * 16;
  int col = lane & 15;
  int row0 = c_row0(lane);

  // X path: s = silu(A @ phix_w1) @ phix_w2 ; X_out = Xn + s*AX
  stage_seq_f16(A, base, S.SA, lane);
  __syncthreads();
  {
    float acc8[8];
#pragma unroll
    for (int v = 0; v < 8; ++v) acc8[v] = 0.0f;
#pragma unroll
    for (int ct = 0; ct < 4; ++ct) {
      v8f c = {0.f, 0.f, 0.f, 0.f, 0.f, 0.f, 0.f, 0.f};
#pragma unroll
      for (int kt = 0; kt < 2; ++kt) {
        v16h a = load_a_frag(S.SA, 64, kt * 32, lane);
        v16h b = load_b_packed(phixw1p, kt * 4 + ct, lane);
        c = wmma16(a, b, c);
      }
      float w2n = phix_w2[ct * 16 + col];
#pragma unroll
      for (int v = 0; v < 8; ++v) acc8[v] += siluf(c[v]) * w2n;
    }
#pragma unroll
    for (int v = 0; v < 8; ++v) S.red[lane * 8 + v] = acc8[v];
  }
  __syncthreads();
  if (lane < 16) {
    int m = lane, v = m & 7, half = m >> 3;
    float s = 0.0f;
#pragma unroll
    for (int j = 0; j < 16; ++j) s += S.red[(half * 16 + j) * 8 + v];
    S.sval[m] = s;
  }
  __syncthreads();
  if (lane < 16) {
    long i = base + lane;
    float s = S.sval[lane];
#pragma unroll
    for (int k = 0; k < 3; ++k) Xout[i * 3 + k] = Xn[i * 3 + k] + s * AX[i * 3 + k];
  }
  __syncthreads();

  // H path: H_out = H + silu((A*AH) @ phih_w1 + b1) @ phih_w2 + b2
  {
    int r = lane & 15, c0 = (lane & 16) ? 32 : 0;
    const float4* a4 = (const float4*)(A + (base + r) * 64 + c0);
    const float4* h4 = (const float4*)(AH + (base + r) * 64 + c0);
    _Float16* dst = S.SA + r * 64 + c0;
#pragma unroll
    for (int j = 0; j < 8; ++j) {
      float4 a = a4[j], h = h4[j];
      float4 m;
      m.x = a.x * h.x; m.y = a.y * h.y; m.z = a.z * h.z; m.w = a.w * h.w;
      cvt_store4(dst + 4 * j, m);
    }
  }
  __syncthreads();
#pragma unroll
  for (int ct = 0; ct < 4; ++ct) {
    v8f c = {0.f, 0.f, 0.f, 0.f, 0.f, 0.f, 0.f, 0.f};
#pragma unroll
    for (int kt = 0; kt < 2; ++kt) {
      v16h a = load_a_frag(S.SA, 64, kt * 32, lane);
      v16h b = load_b_packed(phihw1p, kt * 4 + ct, lane);
      c = wmma16(a, b, c);
    }
    int n = ct * 16 + col;
    float b1n = phih_b1[n];
#pragma unroll
    for (int v = 0; v < 8; ++v) S.SB[(row0 + v) * 64 + n] = (_Float16)siluf(c[v] + b1n);
  }
  __syncthreads();
#pragma unroll
  for (int ct = 0; ct < 4; ++ct) {
    v8f c = {0.f, 0.f, 0.f, 0.f, 0.f, 0.f, 0.f, 0.f};
#pragma unroll
    for (int kt = 0; kt < 2; ++kt) {
      v16h a = load_a_frag(S.SB, 64, kt * 32, lane);
      v16h b = load_b_packed(phihw2p, kt * 4 + ct, lane);
      c = wmma16(a, b, c);
    }
    int n = ct * 16 + col;
    float b2n = phih_b2[n];
#pragma unroll
    for (int v = 0; v < 8; ++v) {
      long i = base + row0 + v;
      Hout[i * 64 + n] = c[v] + b2n + H[i * 64 + n];
    }
  }
}

// -------------------- host launch --------------------

extern "C" void kernel_launch(void* const* d_in, const int* in_sizes, int n_in,
                              void* d_out, int out_size, void* d_ws, size_t ws_size,
                              hipStream_t stream) {
  const int*   batch   = (const int*)d_in[0];
  const float* X       = (const float*)d_in[1];
  const float* H       = (const float*)d_in[2];
  const int*   Eidx    = (const int*)d_in[3];
  const float* ZE      = (const float*)d_in[4];
  const float* kv_w1   = (const float*)d_in[5];
  const float* kv_w2   = (const float*)d_in[6];
  const float* q_w1    = (const float*)d_in[7];
  const float* q_w2    = (const float*)d_in[8];
  const float* pb_w1   = (const float*)d_in[9];
  const float* pb_b1   = (const float*)d_in[10];
  const float* pb_w2   = (const float*)d_in[11];
  const float* pb_b2   = (const float*)d_in[12];
  const float* gate_w1 = (const float*)d_in[13];
  const float* gate_b1 = (const float*)d_in[14];
  const float* gate_w2 = (const float*)d_in[15];
  const float* gate_b2 = (const float*)d_in[16];
  const float* phih_w1 = (const float*)d_in[17];
  const float* phih_b1 = (const float*)d_in[18];
  const float* phih_w2 = (const float*)d_in[19];
  const float* phih_b2 = (const float*)d_in[20];
  const float* phix_w1 = (const float*)d_in[21];
  const float* phix_w2 = (const float*)d_in[22];
  const float* qn_w    = (const float*)d_in[23];
  const float* qn_b    = (const float*)d_in[24];
  const float* kn_w    = (const float*)d_in[25];
  const float* kn_b    = (const float*)d_in[26];
  const float* e3_w    = (const float*)d_in[27];

  float* ws   = (float*)d_ws;
  float* sumn = ws;                 // G
  float* cnt  = ws + GG;            // G
  float* gate = ws + 2 * GG;        // N
  float* Xn   = gate + NN;          // 3N
  float* Aacc = Xn + 3L * NN;       // 64N
  float* AX   = Aacc + 64L * NN;    // 3N
  float* AH   = AX + 3L * NN;       // 64N
  long nFloat = 2L * GG + 135L * NN;

  // packed f16 weights (fragment-major), each a multiple of 512 halves
  _Float16* h16     = (_Float16*)(ws + nFloat);
  _Float16* qw1p    = h16;                    // 64x64   -> 2*4*512  = 4096
  _Float16* qw2p    = qw1p + 4096;            // 4096
  _Float16* kvw1p   = qw2p + 4096;            // 64x128 (rows 1..64) -> 2*8*512 = 8192
  _Float16* kvw2p   = kvw1p + 8192;           // 128x128 -> 4*8*512 = 16384
  _Float16* pbw1p   = kvw2p + 16384;          // 64x256  -> 2*16*512 = 16384
  _Float16* gatew1p = pbw1p + 16384;          // 4096
  _Float16* phixw1p = gatew1p + 4096;         // 4096
  _Float16* phihw1p = phixw1p + 4096;         // 4096
  _Float16* phihw2p = phihw1p + 4096;         // 4096

  // zero accumulators (sumn, cnt, gate, Xn, A, AX, AH)
  zero_f32<<<4096, 256, 0, stream>>>(ws, nFloat);

  // repack weights f32 -> fragment-major f16
  pack_b_f16<<<(4096 + 255) / 256, 256, 0, stream>>>(q_w1, qw1p, 64, 0, 4096);
  pack_b_f16<<<(4096 + 255) / 256, 256, 0, stream>>>(q_w2, qw2p, 64, 0, 4096);
  pack_b_f16<<<(8192 + 255) / 256, 256, 0, stream>>>(kv_w1, kvw1p, 128, 1, 8192);   // skip row 0
  pack_b_f16<<<(16384 + 255) / 256, 256, 0, stream>>>(kv_w2, kvw2p, 128, 0, 16384);
  pack_b_f16<<<(16384 + 255) / 256, 256, 0, stream>>>(pb_w1, pbw1p, 256, 0, 16384);
  pack_b_f16<<<(4096 + 255) / 256, 256, 0, stream>>>(gate_w1, gatew1p, 64, 0, 4096);
  pack_b_f16<<<(4096 + 255) / 256, 256, 0, stream>>>(phix_w1, phixw1p, 64, 0, 4096);
  pack_b_f16<<<(4096 + 255) / 256, 256, 0, stream>>>(phih_w1, phihw1p, 64, 0, 4096);
  pack_b_f16<<<(4096 + 255) / 256, 256, 0, stream>>>(phih_w2, phihw2p, 64, 0, 4096);

  // per-graph mean norm, then Xn
  norm_kernel<<<(NN + 255) / 256, 256, 0, stream>>>(X, batch, sumn, cnt);
  xn_kernel<<<(NN + 255) / 256, 256, 0, stream>>>(X, batch, sumn, cnt, e3_w, Xn);

  // per-node gate (WMMA), N/16 = 3125 tiles
  int ntiles = NN / 16;
  gate_kernel<<<(ntiles + 1) / 2, 64, 0, stream>>>(H, gatew1p, gate_b1, gate_w2, gate_b2, gate, ntiles);

  // per-edge attention (WMMA), E/16 = 50000 tiles, 2 waves/block
  edge_kernel<<<(EE / 16) / 2, 64, 0, stream>>>(
      Eidx, H, Xn, ZE, qw1p, qw2p, kvw1p, kvw2p, pbw1p, kv_w1, pb_b1, pb_w2, pb_b2,
      qn_w, qn_b, kn_w, kn_b, gate, Aacc, AX, AH);

  // per-node outputs (WMMA)
  float* Xout = (float*)d_out;
  float* Hout = Xout + 3L * NN;
  nodeout_kernel<<<(ntiles + 1) / 2, 64, 0, stream>>>(
      H, Xn, Aacc, AX, AH, phixw1p, phix_w2, phihw1p, phih_b1, phihw2p, phih_b2,
      Xout, Hout, ntiles);
}
